// GRU_2052994367725
// MI455X (gfx1250) — matrix operand order repaired and checked by reading, not compiled, not run on hardware
//
#include <hip/hip_runtime.h>
#include <hip/hip_bf16.h>

typedef __attribute__((ext_vector_type(16))) _Float16 v16h;
typedef __attribute__((ext_vector_type(8)))  _Float16 v8h;
typedef __attribute__((ext_vector_type(8)))  float    v8f;

#define NB   256
#define TT   512
#define II   512
#define HH   512
#define G3   1536           // 3*H
#define MROWS (NB * TT)     // 131072 flattened (n,t) rows
#define LDSPAD 8            // pad (in halves) to break LDS bank conflicts

__device__ __forceinline__ v16h cat8(v8h lo, v8h hi) {
    return __builtin_shufflevector(lo, hi, 0,1,2,3,4,5,6,7,8,9,10,11,12,13,14,15);
}

__device__ __forceinline__ v8f wmma_f16(v16h a, v16h b, v8f c) {
    // D = A(16x32) * B(32x16) + C, f32 accumulate
    return __builtin_amdgcn_wmma_f32_16x16x32_f16(false, a, false, b, (short)0, c, false, false);
}

// ---------------------------------------------------------------------------
// Convert weights f32 -> f16 (both W_ih and W_hh, same element count)
// ---------------------------------------------------------------------------
__global__ __launch_bounds__(256)
void k_cvt_w(const float* __restrict__ wih, const float* __restrict__ whh,
             _Float16* __restrict__ wih_h, _Float16* __restrict__ whh_h) {
    size_t i = (size_t)blockIdx.x * 256 + threadIdx.x;   // < G3*II
    wih_h[i] = (_Float16)wih[i];
    whh_h[i] = (_Float16)whh[i];
}

// h0 = hx[:, 0, :]
__global__ __launch_bounds__(256)
void k_init_h(const float* __restrict__ hx, float* __restrict__ hstate) {
    size_t i = (size_t)blockIdx.x * 256 + threadIdx.x;   // < NB*HH
    size_t n = i >> 9, k = i & 511;
    hstate[i] = hx[n * (size_t)TT * HH + k];
}

// ---------------------------------------------------------------------------
// GEMM1: gi[r, g] = f16( sum_k x[r,k]*W_ih[g,k] + b_ih[g] )
// block: 256 thr (8 waves) computes 16 rows x 1536 cols; A staged in LDS.
// ---------------------------------------------------------------------------
__global__ __launch_bounds__(256)
void k_gemm1(const float* __restrict__ x, const _Float16* __restrict__ wih_h,
             const float* __restrict__ b_ih, _Float16* __restrict__ gi) {
    __shared__ _Float16 xs[16][II + LDSPAD];

    const int r0  = blockIdx.x * 16;
    const int tid = threadIdx.x;

    // stage x[r0:r0+16, :] as f16 into LDS
    const float4* xp = (const float4*)(x + (size_t)r0 * II);
    for (int i = tid; i < 16 * II / 4; i += 256) {
        float4 v = xp[i];
        int e = i * 4, m = e >> 9, k = e & 511;
        xs[m][k + 0] = (_Float16)v.x;
        xs[m][k + 1] = (_Float16)v.y;
        xs[m][k + 2] = (_Float16)v.z;
        xs[m][k + 3] = (_Float16)v.w;
    }
    __syncthreads();

    const int wave = tid >> 5, lane = tid & 31;
    const int mrow  = lane & 15;
    const int aoff  = (lane >= 16) ? 16 : 0;   // A: K sub-chunk offset
    const int bcol  = lane & 15;               // B: column within tile
    const int bkoff = (lane >= 16) ? 16 : 0;   // B: K offset
    const int mbase = (lane >= 16) ? 8 : 0;    // C/D: row group

    for (int ti = 0; ti < 12; ++ti) {
        const int cbase = (wave * 12 + ti) * 16;
        v8f acc = {};
        const _Float16* brow = wih_h + (size_t)(cbase + bcol) * II + bkoff;
        #pragma unroll 4
        for (int kk = 0; kk < 16; ++kk) {
            const int ks = kk * 32 + aoff;
            v16h A = cat8(*(const v8h*)&xs[mrow][ks],
                          *(const v8h*)&xs[mrow][ks + 8]);
            const _Float16* bp = brow + kk * 32;
            v16h B = cat8(*(const v8h*)bp, *(const v8h*)(bp + 8));
            acc = wmma_f16(A, B, acc);
        }
        const int   col  = cbase + bcol;
        const float bias = b_ih[col];
        #pragma unroll
        for (int vg = 0; vg < 8; ++vg) {
            int m = mbase + vg;
            gi[(size_t)(r0 + m) * G3 + col] = (_Float16)(acc[vg] + bias);
        }
    }
}

// ---------------------------------------------------------------------------
// Recurrent scan: 16 blocks x 512 thr (16 waves); block owns 16 batch rows,
// loops over all T with LDS-resident h. Per step:
//   [barrier] -> prefetch reset(t+1) + 6 WMMA tile accumulations (reads abuf)
//   [barrier] -> fused gate epilogue writes hbuf (f32 state), abuf (f16
//                operand for t+1, reset pre-applied) and the pre-LN output.
// ---------------------------------------------------------------------------
__global__ __launch_bounds__(512)
void k_scan(const _Float16* __restrict__ gi, const _Float16* __restrict__ whh_h,
            const unsigned char* __restrict__ is_init,
            const float* __restrict__ b_hh,
            float* __restrict__ hstate, float* __restrict__ out) {
    __shared__ float    hbuf[16][HH];
    __shared__ _Float16 abuf[16][HH + LDSPAD];
    __shared__ float    rstbuf[2][16];

    const int n0  = blockIdx.x * 16;
    const int tid = threadIdx.x;

    for (int i = tid; i < 16 * HH; i += 512)
        hbuf[i >> 9][i & 511] = hstate[(size_t)n0 * HH + i];
    if (tid < 16)
        rstbuf[0][tid] = 1.0f - (float)is_init[(size_t)(n0 + tid) * TT + 0];
    __syncthreads();
    // initial A operand: f16(h0 * reset(0))
    for (int i = tid; i < 16 * HH; i += 512) {
        int m = i >> 9, k = i & 511;
        abuf[m][k] = (_Float16)(hbuf[m][k] * rstbuf[0][m]);
    }

    const int wave  = tid >> 5, lane = tid & 31;
    const int mrow  = lane & 15;
    const int aoff  = (lane >= 16) ? 16 : 0;
    const int bcol  = lane & 15;
    const int bkoff = (lane >= 16) ? 16 : 0;
    const int mbase = (lane >= 16) ? 8 : 0;

    for (int t = 0; t < TT; ++t) {
        __syncthreads();            // abuf/hbuf(t) ready for all waves
        const int cur = t & 1, nxt = cur ^ 1;
        if (tid < 16) {
            const int tn = (t + 1 < TT) ? t + 1 : t;
            rstbuf[nxt][tid] =
                1.0f - (float)is_init[(size_t)(n0 + tid) * TT + tn];
        }

        // ---- 6 tile accumulations (2 column sub-tiles x 3 gates) ----
        v8f acc[2][3];
        #pragma unroll
        for (int sub = 0; sub < 2; ++sub) {
            const int c0 = wave * 32 + sub * 16;   // column tile within a gate block
            v8f accr = {}, accz = {}, accn = {};
            const _Float16* br = whh_h + (size_t)(          c0 + bcol) * HH + bkoff;
            const _Float16* bz = whh_h + (size_t)( HH     + c0 + bcol) * HH + bkoff;
            const _Float16* bn = whh_h + (size_t)( 2 * HH + c0 + bcol) * HH + bkoff;
            for (int kk = 0; kk < 16; ++kk) {
                const int ks = kk * 32 + aoff;
                v16h A = cat8(*(const v8h*)&abuf[mrow][ks],
                              *(const v8h*)&abuf[mrow][ks + 8]);
                const _Float16* p;
                p = br + kk * 32; v16h Br = cat8(*(const v8h*)p, *(const v8h*)(p + 8));
                p = bz + kk * 32; v16h Bz = cat8(*(const v8h*)p, *(const v8h*)(p + 8));
                p = bn + kk * 32; v16h Bn = cat8(*(const v8h*)p, *(const v8h*)(p + 8));
                accr = wmma_f16(A, Br, accr);
                accz = wmma_f16(A, Bz, accz);
                accn = wmma_f16(A, Bn, accn);
            }
            acc[sub][0] = accr; acc[sub][1] = accz; acc[sub][2] = accn;
        }

        __syncthreads();            // all wmma reads of abuf(t) complete

        // ---- fused gate epilogue; each lane owns (rows mbase..+7, col jr) ----
        #pragma unroll
        for (int sub = 0; sub < 2; ++sub) {
            const int   jr   = wave * 32 + sub * 16 + bcol;
            const float bh_r = b_hh[jr];
            const float bh_z = b_hh[HH + jr];
            const float bh_n = b_hh[2 * HH + jr];
            #pragma unroll
            for (int vg = 0; vg < 8; ++vg) {
                const int    m  = mbase + vg;
                const size_t rr = (size_t)(n0 + m) * TT + t;
                const _Float16* gp = gi + rr * G3;
                float ir  = (float)gp[jr]          + acc[sub][0][vg] + bh_r;
                float iz  = (float)gp[HH + jr]     + acc[sub][1][vg] + bh_z;
                float inn = (float)gp[2 * HH + jr];
                float hn  = acc[sub][2][vg] + bh_n;
                float r   = 1.0f / (1.0f + __expf(-ir));
                float z   = 1.0f / (1.0f + __expf(-iz));
                float ng  = tanhf(inn + r * hn);
                // reference applies reset to h BEFORE the convex combination
                float ho  = hbuf[m][jr] * rstbuf[cur][m];
                float hne = (1.0f - z) * ng + z * ho;
                hbuf[m][jr]       = hne;
                abuf[m][jr]       = (_Float16)(hne * rstbuf[nxt][m]);
                out[rr * HH + jr] = hne;
            }
        }
        // next iteration's top __syncthreads orders abuf/hbuf writes vs reads
    }
    __syncthreads();
    for (int i = tid; i < 16 * HH; i += 512)
        hstate[(size_t)n0 * HH + i] = hbuf[i >> 9][i & 511];
}

// ---------------------------------------------------------------------------
// In-place LayerNorm over each (n,t) row of the GRU output + h_last broadcast
// ---------------------------------------------------------------------------
__global__ __launch_bounds__(256)
void k_ln(float* __restrict__ out, const float* __restrict__ hstate,
          const float* __restrict__ gamma, const float* __restrict__ beta,
          float* __restrict__ hx_rep) {
    __shared__ float ssum[8], ssq[8];
    const int    row = blockIdx.x;          // 0..MROWS-1
    const int    n   = row >> 9;            // row / T
    const int    tid = threadIdx.x;
    float* o = out + (size_t)row * HH;

    float v0 = o[tid], v1 = o[tid + 256];
    float s  = v0 + v1;
    float s2 = v0 * v0 + v1 * v1;
    const int w = tid >> 5, lane = tid & 31;
    #pragma unroll
    for (int off = 16; off > 0; off >>= 1) {
        s  += __shfl_down(s,  off);
        s2 += __shfl_down(s2, off);
    }
    if (lane == 0) { ssum[w] = s; ssq[w] = s2; }
    __syncthreads();
    if (tid == 0) {
        float a = 0.f, b = 0.f;
        #pragma unroll
        for (int i = 0; i < 8; ++i) { a += ssum[i]; b += ssq[i]; }
        ssum[0] = a; ssq[0] = b;
    }
    __syncthreads();
    const float mu  = ssum[0] * (1.0f / HH);
    const float var = ssq[0] * (1.0f / HH) - mu * mu;
    const float inv = rsqrtf(var + 1e-5f);

    o[tid]       = (v0 - mu) * inv * gamma[tid]       + beta[tid];
    o[tid + 256] = (v1 - mu) * inv * gamma[tid + 256] + beta[tid + 256];

    float* hr = hx_rep + (size_t)row * HH;
    const float* hs = hstate + (size_t)n * HH;
    hr[tid]       = hs[tid];
    hr[tid + 256] = hs[tid + 256];
}

// ---------------------------------------------------------------------------
extern "C" void kernel_launch(void* const* d_in, const int* in_sizes, int n_in,
                              void* d_out, int out_size, void* d_ws, size_t ws_size,
                              hipStream_t stream) {
    (void)in_sizes; (void)n_in; (void)out_size; (void)ws_size;

    const float*         x       = (const float*)d_in[0];
    const unsigned char* is_init = (const unsigned char*)d_in[1];
    const float*         hx      = (const float*)d_in[2];
    const float*         W_ih    = (const float*)d_in[3];
    const float*         W_hh    = (const float*)d_in[4];
    const float*         b_ih    = (const float*)d_in[5];
    const float*         b_hh    = (const float*)d_in[6];
    const float*         gamma   = (const float*)d_in[7];
    const float*         beta    = (const float*)d_in[8];

    float* out_ln = (float*)d_out;
    float* hx_rep = out_ln + (size_t)MROWS * HH;

    char* ws = (char*)d_ws;
    _Float16* wih_h  = (_Float16*)ws;                                     // 1.5 MB
    _Float16* whh_h  = (_Float16*)(ws + (size_t)G3 * II * 2);             // 1.5 MB
    float*    hstate = (float*)  (ws + (size_t)G3 * II * 4);              // 0.5 MB
    _Float16* gi     = (_Float16*)(ws + (size_t)G3 * II * 4
                                      + (size_t)NB * HH * 4);             // 403 MB

    k_cvt_w <<<G3 * II / 256, 256, 0, stream>>>(W_ih, W_hh, wih_h, whh_h);
    k_init_h<<<NB * HH / 256, 256, 0, stream>>>(hx, hstate);
    k_gemm1 <<<MROWS / 16,    256, 0, stream>>>(x, wih_h, b_ih, gi);
    k_scan  <<<NB / 16,       512, 0, stream>>>(gi, whh_h, is_init, b_hh, hstate, out_ln);
    k_ln    <<<MROWS,         256, 0, stream>>>(out_ln, hstate, gamma, beta, hx_rep);
}
